// AttentionOp_76218489634850
// MI455X (gfx1250) — matrix operand order repaired and checked by reading, not compiled, not run on hardware
//
#include <hip/hip_runtime.h>
#include <hip/hip_bf16.h>
#include <math.h>

// ---------------------------------------------------------------------------
// Linear attention block (B=4, L=8192, D=512, H=8, HD=64).
// Big GEMMs (qkv, out-proj): bf16x3-split on V_WMMA_F32_16X16X32_BF16,
// K-outer / N-tile-inner with per-tile accumulators (bounded VGPRs, load/WMMA
// overlap). Small per-head GEMMs: exact V_WMMA_F32_16X16X4_F32.
// ---------------------------------------------------------------------------

#define B_   4
#define L_   8192
#define D_   512
#define H_   8
#define HD_  64
#define M_   (B_ * L_)          // 32768 rows
#define N3_  (3 * D_)           // 1536
#define EPS_NORM 1e-8f
#define EPS_RMS  1e-8f

typedef float  v2f   __attribute__((ext_vector_type(2)));
typedef float  v8f   __attribute__((ext_vector_type(8)));
typedef __bf16 v8bf  __attribute__((ext_vector_type(8)));
typedef __bf16 v16bf __attribute__((ext_vector_type(16)));

__device__ __forceinline__ v8f wmma_f32_k4(v2f a, v2f b, v8f c) {
  return __builtin_amdgcn_wmma_f32_16x16x4_f32(false, a, false, b,
                                               (short)0, c, false, false);
}
__device__ __forceinline__ v8f wmma_bf16(v16bf a, v16bf b, v8f c) {
  return __builtin_amdgcn_wmma_f32_16x16x32_bf16(false, a, false, b,
                                                 (short)0, c, false, false);
}

__device__ __forceinline__ float elu1(float x) {
  return x > 0.0f ? x + 1.0f : __expf(x);   // elu(x)+1
}
__device__ __forceinline__ v8f vzero8() {
  v8f z = {0.f, 0.f, 0.f, 0.f, 0.f, 0.f, 0.f, 0.f};
  return z;
}

// exact hi/lo split: hi = x with mantissa truncated to bf16, lo = bf16(x-hi).
__device__ __forceinline__ void split2(float x, __bf16& hi, __bf16& lo) {
  const float hf = __uint_as_float(__float_as_uint(x) & 0xffff0000u);
  hi = (__bf16)hf;
  lo = (__bf16)(x - hf);
}

// A-fragment (16x32 bf16): lane(half,r) needs K run [k0+16h,+16) of row r
// -> two contiguous b128 reads.
__device__ __forceinline__ v16bf ldA(const __bf16* row, int k0, int half) {
  v8bf a0 = *(const v8bf*)(row + k0 + 16 * half);
  v8bf a1 = *(const v8bf*)(row + k0 + 16 * half + 8);
  return __builtin_shufflevector(a0, a1, 0, 1, 2, 3, 4, 5, 6, 7,
                                 8, 9, 10, 11, 12, 13, 14, 15);
}

// ---------------------------------------------------------------------------
// K0: split a weight matrix into bf16 hi/lo planes (once per launch).
// ---------------------------------------------------------------------------
__global__ __launch_bounds__(256)
void wsplit_kernel(const float* __restrict__ w, __bf16* __restrict__ hi,
                   __bf16* __restrict__ lo, int n) {
  const int i = blockIdx.x * 256 + threadIdx.x;
  if (i < n) { __bf16 h, l; split2(w[i], h, l); hi[i] = h; lo[i] = l; }
}

// ---------------------------------------------------------------------------
// K1: qkv = x @ Wqkv^T  (M=32768, N=1536, K=512) via bf16x3 WMMA.
// K-outer, 12 N-tiles inner; A fragment reused 12x per K-step.
// ---------------------------------------------------------------------------
__global__ __launch_bounds__(256)
void qkv_kernel(const float* __restrict__ x,
                const __bf16* __restrict__ Wh, const __bf16* __restrict__ Wl,
                float* __restrict__ qo, float* __restrict__ ko,
                float* __restrict__ vo) {
  __shared__ __bf16 Ah[16 * D_];                      // 16 KB
  __shared__ __bf16 Al[16 * D_];                      // 16 KB
  const int tid = threadIdx.x;
  const int m0  = blockIdx.x * 16;

  for (int i = tid; i < 16 * D_; i += 256) {          // stage + split A tile
    __bf16 h, l;
    split2(x[(size_t)(m0 + (i >> 9)) * D_ + (i & 511)], h, l);
    Ah[i] = h; Al[i] = l;
  }
  __syncthreads();

  const int wid = tid >> 5, lane = tid & 31;
  const int half = lane >> 4, r = lane & 15;
  const __bf16* arh = &Ah[r * D_];
  const __bf16* arl = &Al[r * D_];
  // tile t covers rows nb = (wid + 8t)*16 of W; stride between tiles is a
  // compile-time 128*D_ elements (128 KB) -> folds into instruction offsets.
  const __bf16* whr = Wh + (size_t)(wid * 16 + r) * D_;
  const __bf16* wlr = Wl + (size_t)(wid * 16 + r) * D_;
  __builtin_prefetch(whr, 0, 1);
  __builtin_prefetch(wlr, 0, 1);

  v8f acc[12];
#pragma unroll
  for (int t = 0; t < 12; ++t) acc[t] = vzero8();

#pragma unroll 1
  for (int k0 = 0; k0 < D_; k0 += 32) {
    const v16bf ah = ldA(arh, k0, half);
    const v16bf al = ldA(arl, k0, half);
#pragma unroll
    for (int t = 0; t < 12; ++t) {
      const size_t toff = (size_t)t * 128 * D_ + k0 + 16 * half;
      const v16bf bh = *(const v16bf*)(whr + toff);
      const v16bf bl = *(const v16bf*)(wlr + toff);
      acc[t] = wmma_bf16(ah, bh, acc[t]);
      acc[t] = wmma_bf16(al, bh, acc[t]);
      acc[t] = wmma_bf16(ah, bl, acc[t]);
    }
  }

#pragma unroll
  for (int t = 0; t < 12; ++t) {
    const int n = (wid + 8 * t) * 16 + r;             // column in [0,1536)
#pragma unroll
    for (int v = 0; v < 8; ++v) {
      const int m = m0 + v + 8 * half;
      float val = acc[t][v];
      if (n < 2 * D_) val = elu1(val);                // feature map on q, k
      if (n < D_)           qo[(size_t)m * D_ + n]            = val;
      else if (n < 2 * D_)  ko[(size_t)m * D_ + (n - D_)]     = val;
      else                  vo[(size_t)m * D_ + (n - 2 * D_)] = val;
    }
  }
}

// ---------------------------------------------------------------------------
// K2: per (b,h): kv = k^T v (64x64, K over L), ksum = sum_l k[l].
// Exact f32 WMMA; L split in 1024-row chunks, global f32 atomics.
// ---------------------------------------------------------------------------
__global__ __launch_bounds__(512)
void kvred_kernel(const float* __restrict__ kf, const float* __restrict__ vf,
                  float* __restrict__ kv, float* __restrict__ ksum) {
  const int chunk = blockIdx.x;                       // 0..7
  const int pair  = blockIdx.y;                       // 0..31 == b*H + h
  const int b = pair >> 3, h = pair & 7;
  const size_t lbase = (size_t)b * L_ + (size_t)chunk * 1024;
  const int col0 = h * HD_;

  const int tid  = threadIdx.x;
  const int wid  = tid >> 5, lane = tid & 31;
  const int half = lane >> 4, r = lane & 15;
  const int db = (wid >> 2) * 16;
  const int mb = (wid & 3) * 16;

  v8f acc = vzero8();
  for (int k0 = 0; k0 < 1024; k0 += 4) {
    const size_t l = lbase + k0 + 2 * half;
    v2f a, bb;
    a.x  = kf[l * D_ + col0 + db + r];
    a.y  = kf[(l + 1) * D_ + col0 + db + r];
    bb.x = vf[l * D_ + col0 + mb + r];
    bb.y = vf[(l + 1) * D_ + col0 + mb + r];
    acc = wmma_f32_k4(a, bb, acc);
  }
  float* kvp = kv + (size_t)pair * (HD_ * HD_);
#pragma unroll
  for (int v = 0; v < 8; ++v)
    atomicAdd(&kvp[(db + v + 8 * half) * HD_ + mb + r], acc[v]);

  {                                                   // ksum: 64 d x 8 groups
    const int d = tid & 63, g = tid >> 6;
    float s = 0.f;
    for (int l = g * 128; l < (g + 1) * 128; ++l)
      s += kf[(lbase + l) * D_ + col0 + d];
    atomicAdd(&ksum[pair * HD_ + d], s);
  }
}

// ---------------------------------------------------------------------------
// K3: out[l] = (q[l] @ kv) / (q[l].ksum + eps), exact f32 WMMA, K=64.
// ---------------------------------------------------------------------------
__global__ __launch_bounds__(128)
void attn_kernel(const float* __restrict__ qf, const float* __restrict__ kv,
                 const float* __restrict__ ksum, float* __restrict__ attn) {
  __shared__ float kvs[HD_ * HD_];
  __shared__ float kss[HD_];
  __shared__ float nrmp[16][8];
  __shared__ float nrm[16];

  const int ltile = blockIdx.x;                       // 0..511
  const int pair  = blockIdx.y;                       // 0..31
  const int b = pair >> 3, h = pair & 7;
  const size_t gl0 = (size_t)b * L_ + (size_t)ltile * 16;
  const int col0 = h * HD_;
  const int tid = threadIdx.x;

  for (int i = tid; i < HD_ * HD_; i += 128)
    kvs[i] = kv[(size_t)pair * (HD_ * HD_) + i];
  if (tid < HD_) kss[tid] = ksum[pair * HD_ + tid];
  __syncthreads();

  {
    const int row = tid >> 3, t = tid & 7;
    const float* qr = qf + (gl0 + row) * D_ + col0;
    float s = 0.f;
#pragma unroll
    for (int j = 0; j < 8; ++j) { const int d = t * 8 + j; s += qr[d] * kss[d]; }
    nrmp[row][t] = s;
  }
  __syncthreads();
  if (tid < 16) {
    float s = 0.f;
#pragma unroll
    for (int j = 0; j < 8; ++j) s += nrmp[tid][j];
    nrm[tid] = s;
  }
  __syncthreads();

  const int wid = tid >> 5, lane = tid & 31;
  const int half = lane >> 4, r = lane & 15;
  const int mb = wid * 16;
  const float* qr = qf + (gl0 + r) * D_ + col0;

  v8f acc = vzero8();
#pragma unroll
  for (int d0 = 0; d0 < HD_; d0 += 4) {
    v2f a = *(const v2f*)(qr + d0 + 2 * half);
    v2f bb;
    bb.x = kvs[(d0 + 2 * half) * HD_ + mb + r];
    bb.y = kvs[(d0 + 2 * half + 1) * HD_ + mb + r];
    acc = wmma_f32_k4(a, bb, acc);
  }
#pragma unroll
  for (int v = 0; v < 8; ++v) {
    const int row = v + 8 * half;
    attn[(gl0 + row) * D_ + col0 + mb + r] = acc[v] / (nrm[row] + EPS_NORM);
  }
}

// ---------------------------------------------------------------------------
// K4: y = attn @ Wout^T + x ; out = scale * y / (rms(y) + eps). bf16x3 WMMA.
// K-outer, 4 N-tiles inner. LDS: bf16 A planes overlaid with f32 Y tile.
// ---------------------------------------------------------------------------
struct SmemK4 {
  union {
    struct { __bf16 hi[16 * D_]; __bf16 lo[16 * D_]; } a;   // 32 KB
    float y[16 * D_];                                        // 32 KB
  } u;
};

__global__ __launch_bounds__(256)
void out_kernel(const float* __restrict__ attn,
                const __bf16* __restrict__ Wh, const __bf16* __restrict__ Wl,
                const float* __restrict__ x, const float* __restrict__ scale,
                float* __restrict__ out) {
  __shared__ SmemK4 sm;
  __shared__ float part[16][16];
  __shared__ float rmsv[16];

  const int tid = threadIdx.x;
  const int m0  = blockIdx.x * 16;

  for (int i = tid; i < 16 * D_; i += 256) {
    __bf16 h, l;
    split2(attn[(size_t)(m0 + (i >> 9)) * D_ + (i & 511)], h, l);
    sm.u.a.hi[i] = h; sm.u.a.lo[i] = l;
  }
  __syncthreads();

  const int wid = tid >> 5, lane = tid & 31;
  const int half = lane >> 4, r = lane & 15;
  const __bf16* arh = &sm.u.a.hi[r * D_];
  const __bf16* arl = &sm.u.a.lo[r * D_];
  const __bf16* whr = Wh + (size_t)(wid * 16 + r) * D_;
  const __bf16* wlr = Wl + (size_t)(wid * 16 + r) * D_;

  v8f acc[4];
#pragma unroll
  for (int t = 0; t < 4; ++t) acc[t] = vzero8();

#pragma unroll 1
  for (int k0 = 0; k0 < D_; k0 += 32) {
    const v16bf ah = ldA(arh, k0, half);
    const v16bf al = ldA(arl, k0, half);
#pragma unroll
    for (int t = 0; t < 4; ++t) {
      const size_t toff = (size_t)t * 128 * D_ + k0 + 16 * half;
      const v16bf bh = *(const v16bf*)(whr + toff);
      const v16bf bl = *(const v16bf*)(wlr + toff);
      acc[t] = wmma_bf16(ah, bh, acc[t]);
      acc[t] = wmma_bf16(al, bh, acc[t]);
      acc[t] = wmma_bf16(ah, bl, acc[t]);
    }
  }
  __syncthreads();                                    // done reading A planes

#pragma unroll
  for (int t = 0; t < 4; ++t) {                       // y = gemm + residual
    const int nb = (wid + 8 * t) * 16;
#pragma unroll
    for (int v = 0; v < 8; ++v) {
      const int row = v + 8 * half;
      const int n = nb + r;
      sm.u.y[row * D_ + n] = acc[t][v] + x[(size_t)(m0 + row) * D_ + n];
    }
  }
  __syncthreads();

  {
    const int row = tid >> 4, c = tid & 15;
    float s = 0.f;
#pragma unroll 8
    for (int j = 0; j < 32; ++j) { const float y = sm.u.y[row * D_ + c + 16 * j]; s += y * y; }
    part[row][c] = s;
  }
  __syncthreads();
  if (tid < 16) {
    float s = 0.f;
#pragma unroll
    for (int j = 0; j < 16; ++j) s += part[tid][j];
    rmsv[tid] = sqrtf(s * (1.0f / (float)D_));
  }
  __syncthreads();

  for (int i = tid; i < 16 * D_; i += 256) {
    const int row = i >> 9, n = i & 511;
    out[(size_t)(m0 + row) * D_ + n] = scale[n] * sm.u.y[i] / (rmsv[row] + EPS_RMS);
  }
}

// ---------------------------------------------------------------------------
extern "C" void kernel_launch(void* const* d_in, const int* in_sizes, int n_in,
                              void* d_out, int out_size, void* d_ws, size_t ws_size,
                              hipStream_t stream) {
  (void)in_sizes; (void)n_in; (void)out_size; (void)ws_size;
  const float* x     = (const float*)d_in[0];
  const float* Wqkv  = (const float*)d_in[1];
  const float* Wout  = (const float*)d_in[2];
  const float* scale = (const float*)d_in[3];
  float* out = (float*)d_out;

  float* ws = (float*)d_ws;
  const size_t MD = (size_t)M_ * D_;                  // 16.78M floats
  float* q_ws    = ws;
  float* k_ws    = q_ws + MD;
  float* v_ws    = k_ws + MD;
  float* attn_ws = v_ws + MD;
  float* kv_ws   = attn_ws + MD;                      // 32*64*64
  float* ksum_ws = kv_ws + (size_t)B_ * H_ * HD_ * HD_;  // 32*64

  const int nWqkv = N3_ * D_;                         // 786432
  const int nWout = D_ * D_;                          // 262144
  __bf16* wqkv_h = (__bf16*)(ksum_ws + (size_t)B_ * H_ * HD_);
  __bf16* wqkv_l = wqkv_h + nWqkv;
  __bf16* wout_h = wqkv_l + nWqkv;
  __bf16* wout_l = wout_h + nWout;

  // zero the cross-chunk accumulators every launch (graph-capture safe)
  hipMemsetAsync(kv_ws, 0,
                 ((size_t)B_ * H_ * HD_ * HD_ + (size_t)B_ * H_ * HD_) * sizeof(float),
                 stream);

  wsplit_kernel<<<(nWqkv + 255) / 256, 256, 0, stream>>>(Wqkv, wqkv_h, wqkv_l, nWqkv);
  wsplit_kernel<<<(nWout + 255) / 256, 256, 0, stream>>>(Wout, wout_h, wout_l, nWout);

  qkv_kernel  <<<M_ / 16, 256, 0, stream>>>(x, wqkv_h, wqkv_l, q_ws, k_ws, v_ws);
  kvred_kernel<<<dim3(L_ / 1024, B_ * H_), 512, 0, stream>>>(k_ws, v_ws, kv_ws, ksum_ws);
  attn_kernel <<<dim3(L_ / 16, B_ * H_), 128, 0, stream>>>(q_ws, kv_ws, ksum_ws, attn_ws);
  out_kernel  <<<M_ / 16, 256, 0, stream>>>(attn_ws, wout_h, wout_l, x, scale, out);
}
